// GatedSparseAttention_47038481826266
// MI455X (gfx1250) — matrix operand order repaired and checked by reading, not compiled, not run on hardware
//
#include <hip/hip_runtime.h>
#include <hip/hip_bf16.h>
#include <math.h>

// ---------------- problem dims ----------------
static constexpr int BB   = 2;
static constexpr int LL   = 2048;
static constexpr int DD   = 1024;
static constexpr int HH   = 16;
static constexpr int HKV  = 4;
static constexpr int DH   = 64;
static constexpr int KSEL = 128;
static constexpr int HI   = 4;
static constexpr int DI   = 64;
static constexpr int NTOK = BB * LL;            // 4096
static constexpr float NEGF = -1e30f;

typedef __bf16 bf16;
typedef __attribute__((ext_vector_type(16))) __bf16 v16bf;
typedef __attribute__((ext_vector_type(8)))  __bf16 v8bf;
typedef __attribute__((ext_vector_type(8)))  float  v8f;
typedef __attribute__((ext_vector_type(4)))  int    v4i;

// ---------------- CDNA5 async-to-LDS availability ----------------
#if defined(__HIP_DEVICE_COMPILE__) && __has_builtin(__builtin_amdgcn_global_load_async_to_lds_b128)
#define HAVE_ASYNC_LDS 1
#else
#define HAVE_ASYNC_LDS 0
#endif

#if HAVE_ASYNC_LDS
typedef __attribute__((address_space(1))) v4i* gv4i_ptr;   // global v4i*
typedef __attribute__((address_space(3))) v4i* lv4i_ptr;   // LDS v4i*

__device__ inline void copy16_async(const bf16* g, bf16* l) {
  __builtin_amdgcn_global_load_async_to_lds_b128(
      (gv4i_ptr)(g), (lv4i_ptr)(l), /*offset=*/0, /*cpol=*/0);
}
__device__ inline void async_commit_barrier() {
#if __has_builtin(__builtin_amdgcn_s_wait_asynccnt)
  __builtin_amdgcn_s_wait_asynccnt(0);
#else
  asm volatile("s_wait_asynccnt 0" ::: "memory");
#endif
  __syncthreads();
}
#else
__device__ inline void async_commit_barrier() { __syncthreads(); }
#endif

// ---------------- WMMA helpers ----------------
__device__ inline v8f wmma_bf16(v16bf a, v16bf b, v8f c) {
  // D = A(16x32 bf16) * B(32x16 bf16) + C(16x16 f32)
  return __builtin_amdgcn_wmma_f32_16x16x32_bf16(
      /*neg_a=*/false, a, /*neg_b=*/false, b,
      /*c_mod=*/(short)0, c, /*reuse_a=*/false, /*reuse_b=*/false);
}

// A fragment: 16x32 (MxK).  lanes 0-15: M=lane, K=kb..kb+15
//             lanes 16-31: M=lane-16, K=kb+16..kb+31
__device__ inline v16bf load_a_frag(const bf16* __restrict__ A, int lda,
                                    int row0, int kb, int lane) {
  int m  = lane & 15;
  int kh = (lane >> 4) & 1;
  const bf16* p = A + (size_t)(row0 + m) * lda + kb + kh * 16;
  v8bf lo = *(const v8bf*)(p);
  v8bf hi = *(const v8bf*)(p + 8);
  v16bf r;
#pragma unroll
  for (int i = 0; i < 8; ++i) { r[i] = lo[i]; r[i + 8] = hi[i]; }
  return r;
}

// B fragment from transposed operand Bt[N][K] (row-major):
// lanes 0-15: N=lane, K=kb..kb+15 ; lanes 16-31: N=lane-16, K=kb+16..kb+31
__device__ inline v16bf load_b_frag(const bf16* __restrict__ Bt, int ldb,
                                    int col0, int kb, int lane) {
  int n  = lane & 15;
  int kh = (lane >> 4) & 1;
  return *(const v16bf*)(Bt + (size_t)(col0 + n) * ldb + kb + kh * 16);
}

// ---------------- conversion kernels ----------------
__global__ void k_convert_bf16(const float* __restrict__ src,
                               bf16* __restrict__ dst, int n) {
  int i = blockIdx.x * blockDim.x + threadIdx.x;
  if (i < n) dst[i] = (bf16)src[i];
}

// dst[n*K + k] = src[k*N + n]   (weights -> transposed bf16)
__global__ void k_transpose_bf16(const float* __restrict__ src,
                                 bf16* __restrict__ dst, int K, int N) {
  int i = blockIdx.x * blockDim.x + threadIdx.x;
  if (i < K * N) {
    int k = i / N, n = i % N;
    dst[(size_t)n * K + k] = (bf16)src[i];
  }
}

// ---------------- NT GEMM: C[M,N] = A[M,K] * Bt[N,K]^T ----------------
// Block tile 32(M) x 256(N), K-step 32, LDS double-buffered (async-to-LDS
// when available).  8 waves; each wave owns a 16x64 register tile
// (4 x 16x16 f32 accumulators -> 4 WMMAs per A fragment).
// EPI: 0 = f32 out, 1 = sigmoid(x + bias[n]) f32 out, 2 = bf16 out
static constexpr int GBM = 32, GBN = 256, GBK = 32;

__device__ inline void stage_tiles(const bf16* __restrict__ A,
                                   const bf16* __restrict__ Bt,
                                   int N, int Kd, int rowblk, int colblk, int kb,
                                   bf16* As, bf16* Bs, int t) {
  // A tile: 32 rows x 32 K = 128 chunks of 8 bf16 (16B)
  if (t < 128) {
    int r = t >> 2, c = (t & 3) * 8;
    const bf16* src = A + (size_t)(rowblk + r) * Kd + kb + c;
    bf16* dst = As + r * GBK + c;
#if HAVE_ASYNC_LDS
    copy16_async(src, dst);
#else
    *(v8bf*)dst = *(const v8bf*)src;
#endif
  }
  // B tile: 256 rows(n) x 32 K = 1024 chunks; 4 per thread
#pragma unroll
  for (int i = 0; i < 4; ++i) {
    int cc = t + i * 256;
    int n = cc >> 2, c = (cc & 3) * 8;
    if (colblk + n < N) {
      const bf16* src = Bt + (size_t)(colblk + n) * Kd + kb + c;
      bf16* dst = Bs + n * GBK + c;
#if HAVE_ASYNC_LDS
      copy16_async(src, dst);
#else
      *(v8bf*)dst = *(const v8bf*)src;
#endif
    }
  }
}

template <int EPI>
__global__ __launch_bounds__(256)
void k_gemm_nt(const bf16* __restrict__ A, const bf16* __restrict__ Bt,
               const float* __restrict__ bias,
               float* __restrict__ Cf, bf16* __restrict__ Cb,
               int M, int N, int Kd) {
  __shared__ bf16 As[2][GBM * GBK];   // 2 x 2KB
  __shared__ bf16 Bs[2][GBN * GBK];   // 2 x 16KB
  const int t    = threadIdx.x;
  const int lane = t & 31;
  const int wave = t >> 5;                       // 0..7
  const int wrow = wave >> 2;                    // 0..1 -> 16-row half
  const int wcol = wave & 3;                     // 0..3 -> 64-col group
  const int rowblk = blockIdx.y * GBM;
  const int colblk = blockIdx.x * GBN;
  const int row0 = wrow * 16;

  v8f acc[4] = {};
  const int nk = Kd / GBK;

  stage_tiles(A, Bt, N, Kd, rowblk, colblk, 0, As[0], Bs[0], t);
  async_commit_barrier();

  for (int kbi = 0; kbi < nk; ++kbi) {
    const int cur = kbi & 1;
    if (kbi + 1 < nk)
      stage_tiles(A, Bt, N, Kd, rowblk, colblk, (kbi + 1) * GBK,
                  As[cur ^ 1], Bs[cur ^ 1], t);
    v16bf a = load_a_frag(As[cur], GBK, row0, 0, lane);
#pragma unroll
    for (int j = 0; j < 4; ++j) {
      v16bf b = load_b_frag(Bs[cur], GBK, wcol * 64 + j * 16, 0, lane);
      acc[j] = wmma_bf16(a, b, acc[j]);
    }
    if (kbi + 1 < nk) async_commit_barrier();
  }

  const int mglob = rowblk + row0 + ((lane >> 4) & 1) * 8;
#pragma unroll
  for (int j = 0; j < 4; ++j) {
    const int ncol = colblk + wcol * 64 + j * 16 + (lane & 15);
    if (ncol < N) {
#pragma unroll
      for (int r = 0; r < 8; ++r) {
        float v = acc[j][r];
        size_t off = (size_t)(mglob + r) * N + ncol;
        if (EPI == 1) {
          Cf[off] = 1.0f / (1.0f + expf(-(v + bias[ncol])));
        } else if (EPI == 2) {
          Cb[off] = (bf16)v;
        } else {
          Cf[off] = v;
        }
      }
    }
  }
  (void)M;
}

// ---------------- elementwise v *= gate ----------------
__global__ void k_mul(float* __restrict__ a, const float* __restrict__ b, int n) {
  int i = blockIdx.x * blockDim.x + threadIdx.x;
  if (i < n) a[i] *= b[i];
}

// ---------------- RoPE on q (H heads) and k (HKV heads) ----------------
__global__ void k_rope(float* __restrict__ qf, float* __restrict__ kf) {
  const int tok = blockIdx.x;
  const int l   = tok & (LL - 1);
  const int t   = threadIdx.x;                     // 256 threads
  for (int p = t; p < HH * 32; p += 256) {
    int hd = p >> 5, i = p & 31;
    float freq = (float)l * powf(10000.0f, -(2.0f * i) / 64.0f);
    float c = cosf(freq), s = sinf(freq);
    float* base = qf + ((size_t)tok * HH + hd) * DH;
    float x1 = base[i], x2 = base[i + 32];
    base[i]      = x1 * c - x2 * s;
    base[i + 32] = x2 * c + x1 * s;
  }
  for (int p = t; p < HKV * 32; p += 256) {
    int hd = p >> 5, i = p & 31;
    float freq = (float)l * powf(10000.0f, -(2.0f * i) / 64.0f);
    float c = cosf(freq), s = sinf(freq);
    float* base = kf + ((size_t)tok * HKV + hd) * DH;
    float x1 = base[i], x2 = base[i + 32];
    base[i]      = x1 * c - x2 * s;
    base[i + 32] = x2 * c + x1 * s;
  }
}

// ---------------- lightning indexer scores (WMMA) ----------------
// scores[b][q][k] = sum_h wts[b,q,h]*sigmoid(qi . ki / 8), causal-masked
__global__ __launch_bounds__(128)
void k_indexer(const bf16* __restrict__ qib, const bf16* __restrict__ kib,
               const float* __restrict__ wts, float* __restrict__ scores) {
  const int lane = threadIdx.x & 31;
  const int wave = threadIdx.x >> 5;               // 0..3
  const int b  = blockIdx.z;
  const int q0 = blockIdx.y * 16;
  const int k0 = (blockIdx.x * 4 + wave) * 16;
  const int n    = lane & 15;
  const int hi16 = (lane >> 4) & 1;

  if (k0 > q0 + 15) {                              // whole tile non-causal
#pragma unroll
    for (int r = 0; r < 8; ++r) {
      int m = r + hi16 * 8;
      scores[((size_t)b * LL + q0 + m) * LL + k0 + n] = NEGF;
    }
    return;
  }

  const bf16* Aq = qib + (size_t)b * LL * (HI * DI);
  const bf16* Bk = kib + (size_t)b * LL * (HI * DI);
  v8f sacc = {};
#pragma unroll
  for (int h = 0; h < HI; ++h) {
    v8f acc = {};
#pragma unroll
    for (int kb = 0; kb < DI; kb += 32) {
      v16bf a  = load_a_frag(Aq, HI * DI, q0, h * DI + kb, lane);
      v16bf bf = load_b_frag(Bk, HI * DI, k0, h * DI + kb, lane);
      acc = wmma_bf16(a, bf, acc);
    }
#pragma unroll
    for (int r = 0; r < 8; ++r) {
      int m = r + hi16 * 8;
      float w  = wts[((size_t)b * LL + q0 + m) * HI + h];
      float sg = 1.0f / (1.0f + expf(-acc[r] * 0.125f));   // 1/sqrt(DI)=1/8
      sacc[r] += w * sg;
    }
  }
#pragma unroll
  for (int r = 0; r < 8; ++r) {
    int m  = r + hi16 * 8;
    int kk = k0 + n;
    float val = (kk <= q0 + m) ? sacc[r] : NEGF;
    scores[((size_t)b * LL + q0 + m) * LL + kk] = val;
  }
}

// ---------------- top-128 per query row (iterative argmax) ----------------
__global__ __launch_bounds__(256)
void k_topk(const float* __restrict__ scores, int* __restrict__ tidx,
            float* __restrict__ tval) {
  __shared__ float svals[LL];
  __shared__ float rmax[256];
  __shared__ int   ridx[256];
  const int t = threadIdx.x;
  const size_t row = blockIdx.x;                   // b*L + q
  const float* src = scores + row * LL;
  for (int i = t; i < LL; i += 256) svals[i] = src[i];
  __syncthreads();
  for (int j = 0; j < KSEL; ++j) {
    float best = -3.4e38f; int bi = 0;
    for (int i = t; i < LL; i += 256) {
      float v = svals[i];
      if (v > best) { best = v; bi = i; }
    }
    rmax[t] = best; ridx[t] = bi;
    __syncthreads();
    for (int s = 128; s > 0; s >>= 1) {
      if (t < s && rmax[t + s] > rmax[t]) { rmax[t] = rmax[t + s]; ridx[t] = ridx[t + s]; }
      __syncthreads();
    }
    if (t == 0) {
      tidx[row * KSEL + j] = ridx[0];
      tval[row * KSEL + j] = rmax[0];
      svals[ridx[0]] = -3.0e38f;                   // below NEGF, removed
    }
    __syncthreads();
  }
}

// ---------------- sparse GQA attention over selected keys ----------------
__global__ __launch_bounds__(128)
void k_sparse_attn(const float* __restrict__ qf, const float* __restrict__ kf,
                   const float* __restrict__ vf, const int* __restrict__ tidx,
                   const float* __restrict__ tval, float* __restrict__ attnout) {
  __shared__ float qsh[DH];
  __shared__ float p[KSEL];
  __shared__ int   idxs[KSEL];
  __shared__ float red[128];
  const int t = threadIdx.x;                       // key slot 0..127
  const size_t bq = blockIdx.x;                    // b*L + q
  const int hkv = blockIdx.y;
  const int b = (int)(bq >> 11);                   // bq / LL

  const int idx = tidx[bq * KSEL + t];
  const bool valid = tval[bq * KSEL + t] > (NEGF * 0.5f);
  idxs[t] = idx;
  const size_t ktok = (size_t)b * LL + idx;
  const float* krow = kf + (ktok * HKV + hkv) * DH;

  for (int rr = 0; rr < HH / HKV; ++rr) {
    const int hq = hkv * (HH / HKV) + rr;
    if (t < DH) qsh[t] = qf[(bq * HH + hq) * DH + t];
    __syncthreads();
    float s = 0.0f;
#pragma unroll 8
    for (int d = 0; d < DH; ++d) s += qsh[d] * krow[d];
    s *= 0.125f;                                   // 1/sqrt(DH)
    s = valid ? s : NEGF;
    red[t] = s; __syncthreads();
    for (int st = 64; st > 0; st >>= 1) {
      if (t < st) red[t] = fmaxf(red[t], red[t + st]);
      __syncthreads();
    }
    float mx = red[0]; __syncthreads();
    float e = expf(s - mx);
    red[t] = e; __syncthreads();
    for (int st = 64; st > 0; st >>= 1) {
      if (t < st) red[t] += red[t + st];
      __syncthreads();
    }
    float sum = red[0]; __syncthreads();
    p[t] = valid ? (e / sum) : 0.0f;
    __syncthreads();
    if (t < DH) {
      float o = 0.0f;
      for (int kk = 0; kk < KSEL; ++kk) {
        size_t vtok = (size_t)b * LL + idxs[kk];
        o += p[kk] * vf[(vtok * HKV + hkv) * DH + t];
      }
      attnout[(bq * HH + hq) * DH + t] = o;
    }
    __syncthreads();
  }
}

// ---------------- G1 gate * attn, convert to bf16 ----------------
__global__ void k_gate_bf16(const float* __restrict__ attn,
                            const float* __restrict__ og,
                            bf16* __restrict__ out, int n) {
  int i = blockIdx.x * blockDim.x + threadIdx.x;
  if (i < n) out[i] = (bf16)(attn[i] * og[i]);
}

// ---------------- host orchestration ----------------
extern "C" void kernel_launch(void* const* d_in, const int* in_sizes, int n_in,
                              void* d_out, int out_size, void* d_ws, size_t ws_size,
                              hipStream_t stream) {
  const float* h    = (const float*)d_in[0];
  const float* wq   = (const float*)d_in[1];
  const float* wk   = (const float*)d_in[2];
  const float* wv   = (const float*)d_in[3];
  const float* wo   = (const float*)d_in[4];
  const float* wiq  = (const float*)d_in[5];
  const float* wik  = (const float*)d_in[6];
  const float* wiw  = (const float*)d_in[7];
  const float* wvg  = (const float*)d_in[8];
  const float* bvg  = (const float*)d_in[9];
  const float* wog  = (const float*)d_in[10];
  const float* bog  = (const float*)d_in[11];

  char* w = (char*)d_ws;
  auto carve = [&](size_t bytes) -> char* {
    char* p = w;
    w += (bytes + 255) & ~(size_t)255;
    return p;
  };
  bf16*  hb    = (bf16*) carve((size_t)NTOK * DD * 2);
  bf16*  wqt   = (bf16*) carve((size_t)1024 * 1024 * 2);
  bf16*  wkt   = (bf16*) carve((size_t)256  * 1024 * 2);
  bf16*  wvt   = (bf16*) carve((size_t)256  * 1024 * 2);
  bf16*  wvgt  = (bf16*) carve((size_t)256  * 1024 * 2);
  bf16*  wiqt  = (bf16*) carve((size_t)256  * 1024 * 2);
  bf16*  wikt  = (bf16*) carve((size_t)256  * 1024 * 2);
  bf16*  wiwt  = (bf16*) carve((size_t)HI   * 1024 * 2);
  bf16*  wogt  = (bf16*) carve((size_t)1024 * 1024 * 2);
  bf16*  wot   = (bf16*) carve((size_t)1024 * 1024 * 2);
  float* qf    = (float*)carve((size_t)NTOK * HH  * DH * 4);
  float* kf    = (float*)carve((size_t)NTOK * HKV * DH * 4);
  float* vf    = (float*)carve((size_t)NTOK * HKV * DH * 4);
  float* vgf   = (float*)carve((size_t)NTOK * HKV * DH * 4);
  bf16*  qib   = (bf16*) carve((size_t)NTOK * HI  * DI * 2);
  bf16*  kib   = (bf16*) carve((size_t)NTOK * HI  * DI * 2);
  float* wtsf  = (float*)carve((size_t)NTOK * HI  * 4);
  float* ogf   = (float*)carve((size_t)NTOK * HH  * DH * 4);
  float* scores= (float*)carve((size_t)BB * LL * LL * 4);
  int*   tidx  = (int*)  carve((size_t)NTOK * KSEL * 4);
  float* tval  = (float*)carve((size_t)NTOK * KSEL * 4);
  float* attnf = (float*)carve((size_t)NTOK * HH * DH * 4);
  bf16*  obf   = (bf16*) carve((size_t)NTOK * HH * DH * 2);

  const int TPB = 256;
  auto cdiv = [](int a, int b) { return (a + b - 1) / b; };

  // 1) convert activations + transpose-convert weights to bf16
  k_convert_bf16<<<cdiv(NTOK * DD, TPB), TPB, 0, stream>>>(h, hb, NTOK * DD);
  k_transpose_bf16<<<cdiv(1024 * 1024, TPB), TPB, 0, stream>>>(wq,  wqt,  1024, 1024);
  k_transpose_bf16<<<cdiv(1024 * 256,  TPB), TPB, 0, stream>>>(wk,  wkt,  1024, 256);
  k_transpose_bf16<<<cdiv(1024 * 256,  TPB), TPB, 0, stream>>>(wv,  wvt,  1024, 256);
  k_transpose_bf16<<<cdiv(1024 * 256,  TPB), TPB, 0, stream>>>(wvg, wvgt, 1024, 256);
  k_transpose_bf16<<<cdiv(1024 * 256,  TPB), TPB, 0, stream>>>(wiq, wiqt, 1024, 256);
  k_transpose_bf16<<<cdiv(1024 * 256,  TPB), TPB, 0, stream>>>(wik, wikt, 1024, 256);
  k_transpose_bf16<<<cdiv(1024 * HI,   TPB), TPB, 0, stream>>>(wiw, wiwt, 1024, HI);
  k_transpose_bf16<<<cdiv(1024 * 1024, TPB), TPB, 0, stream>>>(wog, wogt, 1024, 1024);
  k_transpose_bf16<<<cdiv(1024 * 1024, TPB), TPB, 0, stream>>>(wo,  wot,  1024, 1024);

  // 2) projection GEMMs (M = NTOK rows of tokens), block tile 32x256
  auto ggrid = [&](int N) { return dim3((unsigned)cdiv(N, GBN), (unsigned)(NTOK / GBM)); };
  k_gemm_nt<0><<<ggrid(1024), TPB, 0, stream>>>(hb, wqt,  nullptr, qf,   nullptr, NTOK, 1024, DD);
  k_gemm_nt<0><<<ggrid(256),  TPB, 0, stream>>>(hb, wkt,  nullptr, kf,   nullptr, NTOK, 256,  DD);
  k_gemm_nt<0><<<ggrid(256),  TPB, 0, stream>>>(hb, wvt,  nullptr, vf,   nullptr, NTOK, 256,  DD);
  k_gemm_nt<1><<<ggrid(256),  TPB, 0, stream>>>(hb, wvgt, bvg,     vgf,  nullptr, NTOK, 256,  DD);
  k_gemm_nt<2><<<ggrid(256),  TPB, 0, stream>>>(hb, wiqt, nullptr, nullptr, qib,  NTOK, 256,  DD);
  k_gemm_nt<2><<<ggrid(256),  TPB, 0, stream>>>(hb, wikt, nullptr, nullptr, kib,  NTOK, 256,  DD);
  k_gemm_nt<0><<<ggrid(HI),   TPB, 0, stream>>>(hb, wiwt, nullptr, wtsf, nullptr, NTOK, HI,   DD);
  k_gemm_nt<1><<<ggrid(1024), TPB, 0, stream>>>(hb, wogt, bog,     ogf,  nullptr, NTOK, 1024, DD);

  // 3) value gate + RoPE
  k_mul<<<cdiv(NTOK * HKV * DH, TPB), TPB, 0, stream>>>(vf, vgf, NTOK * HKV * DH);
  k_rope<<<NTOK, TPB, 0, stream>>>(qf, kf);

  // 4) indexer scores (WMMA) + top-k
  k_indexer<<<dim3(LL / 64, LL / 16, BB), 128, 0, stream>>>(qib, kib, wtsf, scores);
  k_topk<<<NTOK, TPB, 0, stream>>>(scores, tidx, tval);

  // 5) sparse attention
  k_sparse_attn<<<dim3(NTOK, HKV), 128, 0, stream>>>(qf, kf, vf, tidx, tval, attnf);

  // 6) output gate -> bf16, final GEMM into d_out
  k_gate_bf16<<<cdiv(NTOK * HH * DH, TPB), TPB, 0, stream>>>(attnf, ogf, obf, NTOK * HH * DH);
  k_gemm_nt<0><<<ggrid(1024), TPB, 0, stream>>>(obf, wot, nullptr, (float*)d_out, nullptr,
                                                NTOK, 1024, HH * DH);
  (void)in_sizes; (void)n_in; (void)out_size; (void)ws_size;
}
